// LocalAttentionBlock_86655260164511
// MI455X (gfx1250) — matrix operand order repaired and checked by reading, not compiled, not run on hardware
//
#include <hip/hip_runtime.h>

#define DEVINL __device__ __forceinline__

typedef _Float16 v16h  __attribute__((ext_vector_type(16)));
typedef _Float16 v8h   __attribute__((ext_vector_type(8)));
typedef float    v8f   __attribute__((ext_vector_type(8)));
typedef float    f32x4 __attribute__((ext_vector_type(4)));

// Problem constants (fixed by the reference)
constexpr int PTS  = 4 * 16384;   // B*S
constexpr int KNB  = 16;          // neighbors
constexpr int C    = 64;          // channels
constexpr float EPS = 1e-5f;

constexpr int NW      = 4;            // waves per block (128 threads)
constexpr int NFRAG   = 50;           // weight A-fragments resident in LDS
constexpr int FRAG_H  = NFRAG * 512;  // halves (32 lanes x 16 per fragment)
constexpr int OSTR    = 72;           // f16 staging stride (16B aligned, conflict-free)
constexpr int GROUPS  = PTS / 16;     // each wave owns groups of 16 points

// Fragment ids: Wq 0..7, Wk 8..15, Wv 16..23, Wpe2 24..31, Wproj 32..39,
//               Wpe1 40..43 (K padded 3->32), Wa1 44..45, Wa2 46..49 (K padded 16->32)

// ---------- small helpers -------------------------------------------------
DEVINL v8f zero8() {
  v8f r;
#pragma unroll
  for (int q = 0; q < 8; ++q) r[q] = 0.f;
  return r;
}
DEVINL v16h zeroh() {
  v16h r;
#pragma unroll
  for (int j = 0; j < 16; ++j) r[j] = (_Float16)0.f;
  return r;
}

DEVINL v8f wmma(v16h a, v16h b, v8f c) {
  return __builtin_amdgcn_wmma_f32_16x16x32_f16(false, a, false, b, (short)0, c,
                                                false, false);
}

// A-matrix 16-bit layout: lane j-slot -> K = (j<8 ? j : j+8) + (hiLane ? 8 : 0)
DEVINL int kidxA(int j) { return (j < 8) ? j : j + 8; }

DEVINL v16h ldfrag(const _Float16* fr, int f, int lane) {
  return *(const v16h*)(fr + (f << 9) + (lane << 4));   // 32B aligned
}

DEVINL v8f ld8(const float* p) {
  f32x4 a = *(const f32x4*)p, b = *(const f32x4*)(p + 4);
  v8f r;
#pragma unroll
  for (int q = 0; q < 4; ++q) { r[q] = a[q]; r[q + 4] = b[q]; }
  return r;
}
DEVINL void st8(float* p, const v8f& v) {
  f32x4 a, b;
#pragma unroll
  for (int q = 0; q < 4; ++q) { a[q] = v[q]; b[q] = v[q + 4]; }
  *(f32x4*)p = a;
  *(f32x4*)(p + 4) = b;
}

// B fragment (K x 16, lanes = N): lo lanes K=0..15, hi lanes K=16..31, +32*kh.
// Source row is row-major with >=64 contiguous contraction values.
DEVINL v16h bfrag_row(const float* rp, bool hi, int kh) {
  const float* p = rp + 32 * kh + (hi ? 16 : 0);
  f32x4 a = *(const f32x4*)(p), b = *(const f32x4*)(p + 4);
  f32x4 c = *(const f32x4*)(p + 8), d = *(const f32x4*)(p + 12);
  v16h r;
#pragma unroll
  for (int q = 0; q < 4; ++q) {
    r[q]      = (_Float16)a[q];
    r[q + 4]  = (_Float16)b[q];
    r[q + 8]  = (_Float16)c[q];
    r[q + 12] = (_Float16)d[q];
  }
  return r;
}
DEVINL v16h bfrag_rowh(const _Float16* rp, bool hi, int kh) {
  const _Float16* p = rp + 32 * kh + (hi ? 16 : 0);      // 16B aligned
  v8h a = *(const v8h*)p, b = *(const v8h*)(p + 8);
  v16h r;
#pragma unroll
  for (int j = 0; j < 8; ++j) { r[j] = a[j]; r[j + 8] = b[j]; }
  return r;
}

// exchange a C/D tile with the partner half-wave (lane ^ 16)
DEVINL v8f xchg16(const v8f& v) {
  v8f r;
#pragma unroll
  for (int q = 0; q < 8; ++q) r[q] = __shfl_xor(v[q], 16, 32);
  return r;
}
// Pack two C/D f32 tiles (rows 16t..16t+15 and 16t+16..) into one B fragment
// covering 32 contraction rows.  Only one cross-half exchange needed.
DEVINL v16h packB_cd(const v8f& te, const v8f& to, bool hi) {
  v8f xe = xchg16(te), xo = xchg16(to);
  v16h d;
#pragma unroll
  for (int j = 0; j < 8; ++j) {
    d[j]     = (_Float16)(hi ? xo[j] : te[j]);
    d[j + 8] = (_Float16)(hi ? to[j] : xe[j]);
  }
  return d;
}

DEVINL float rmax16(float x) {
#pragma unroll
  for (int m = 1; m <= 8; m <<= 1) x = fmaxf(x, __shfl_xor(x, m, 32));
  return x;
}
DEVINL float rsum16(float x) {
#pragma unroll
  for (int m = 1; m <= 8; m <<= 1) x += __shfl_xor(x, m, 32);
  return x;
}

DEVINL float pick(const v8f& v, int r) {
  float x = v[0];
#pragma unroll
  for (int q = 1; q < 8; ++q)
    if (r == q) x = v[q];
  return x;
}

// ---------- fused kernel --------------------------------------------------
__global__ __launch_bounds__(NW * 32, 1)
void pt_attn_fused(const float* __restrict__ cxyz,  const float* __restrict__ cfeat,
                   const float* __restrict__ nxyz,  const float* __restrict__ nfeat,
                   const float* __restrict__ Wq,    const float* __restrict__ Wk,
                   const float* __restrict__ Wv,    const float* __restrict__ Wpe1,
                   const float* __restrict__ Wpe2,  const float* __restrict__ Wa1,
                   const float* __restrict__ Wa2,   const float* __restrict__ Wproj,
                   const float* __restrict__ lnw,   const float* __restrict__ lnb,
                   float* __restrict__ out)
{
  extern __shared__ char smem[];
  _Float16* frag   = (_Float16*)smem;                        // 50 KB A-fragments
  _Float16* stoAll = (_Float16*)(smem + FRAG_H * 2);         // per-wave f16 staging

  const int lane = threadIdx.x & 31;
  const int wv_  = threadIdx.x >> 5;
  const bool hi  = lane >= 16;
  const int  ln  = lane & 15;
  const int  hi8 = hi ? 8 : 0;
  _Float16* sto = stoAll + wv_ * (16 * OSTR);

  // ---- build weight A-fragments in LDS (f16, exact WMMA lane layout) ----
  {
    const float* mats[8] = {Wq, Wk, Wv, Wpe2, Wproj, Wpe1, Wa1, Wa2};
    for (int e = threadIdx.x; e < FRAG_H; e += blockDim.x) {
      int f = e >> 9, idx = e & 511;
      int la = idx >> 4, j = idx & 15;
      int lhi8 = (la >= 16) ? 8 : 0, lln = la & 15;
      int m, t, kh, ncols;
      if (f < 40)      { m = f >> 3; t = (f >> 1) & 3; kh = f & 1; ncols = 64; }
      else if (f < 44) { m = 5; t = f - 40; kh = 0;      ncols = 3;  }
      else if (f < 46) { m = 6; t = 0;      kh = f - 44; ncols = 64; }
      else             { m = 7; t = f - 46; kh = 0;      ncols = 16; }
      int row = 16 * t + lln;
      int col = j + 2 * lhi8 + 32 * kh;
      float v = (col < ncols) ? mats[m][row * ncols + col] : 0.0f;
      frag[e] = (_Float16)v;
    }
  }
  __syncthreads();

  const int gw  = blockIdx.x * NW + wv_;
  const int ngw = gridDim.x * NW;

  for (int g = gw; g < GROUPS; g += ngw) {
    const int p0 = g * 16;
    int gn = g + ngw;                 // prefetch next group's neighbor features
    if (gn < GROUPS)
      __builtin_prefetch(nfeat + (size_t)gn * 16 * KNB * C + lane * 32, 0, 0);

    // ---- Phase A: qT = Wq @ cfT for the 16 points (lanes = point) --------
    v8f qacc[4];
    {
      const float* rp = cfeat + (size_t)(p0 + ln) * C;
      v16h b0 = bfrag_row(rp, hi, 0);
      v16h b1 = bfrag_row(rp, hi, 1);
#pragma unroll
      for (int t = 0; t < 4; ++t) {
        v8f a = wmma(ldfrag(frag, 2 * t, lane), b0, zero8());
        qacc[t] = wmma(ldfrag(frag, 2 * t + 1, lane), b1, a);
      }
    }

    // ---- Phase B: per-point attention (lanes = neighbor) -----------------
#pragma unroll 1
    for (int i = 0; i < 16; ++i) {
      const size_t p = (size_t)(p0 + i);

      // relative-position B fragment: K=0..2 valid on lo lanes only
      v16h brel = zeroh();
      if (!hi) {
        const float* nx = nxyz + (p * KNB + ln) * 3;
        const float* cx = cxyz + p * 3;
        brel[0] = (_Float16)(nx[0] - cx[0]);
        brel[1] = (_Float16)(nx[1] - cx[1]);
        brel[2] = (_Float16)(nx[2] - cx[2]);
      }
      // pe1 = relu(Wpe1 @ relT) : 64 x 16
      v8f pe1[4];
#pragma unroll
      for (int t = 0; t < 4; ++t) {
        pe1[t] = wmma(ldfrag(frag, 40 + t, lane), brel, zero8());
#pragma unroll
        for (int r = 0; r < 8; ++r) pe1[t][r] = fmaxf(pe1[t][r], 0.f);
      }
      v16h bpe0 = packB_cd(pe1[0], pe1[1], hi);
      v16h bpe1 = packB_cd(pe1[2], pe1[3], hi);

      // neighbor-feature B fragments (per-lane contiguous rows)
      const float* nf = nfeat + (p * KNB + ln) * C;
      v16h bnf0 = bfrag_row(nf, hi, 0);
      v16h bnf1 = bfrag_row(nf, hi, 1);

      // per channel-tile: pe, kk, v ; build s = q - kk + pe and vpe = v + pe
      v16h bs0, bs1;
      v8f vpe[4];
#pragma unroll
      for (int hh = 0; hh < 2; ++hh) {
        v8f se[2];
#pragma unroll
        for (int u = 0; u < 2; ++u) {
          const int t = 2 * hh + u;
          v8f pe_t = wmma(ldfrag(frag, 24 + 2 * t, lane), bpe0, zero8());
          pe_t     = wmma(ldfrag(frag, 25 + 2 * t, lane), bpe1, pe_t);
          v8f kk_t = wmma(ldfrag(frag,  8 + 2 * t, lane), bnf0, zero8());
          kk_t     = wmma(ldfrag(frag,  9 + 2 * t, lane), bnf1, kk_t);
          v8f vv_t = wmma(ldfrag(frag, 16 + 2 * t, lane), bnf0, zero8());
          vv_t     = wmma(ldfrag(frag, 17 + 2 * t, lane), bnf1, vv_t);
#pragma unroll
          for (int r = 0; r < 8; ++r) {
            float qv = __shfl(qacc[t][r], i | (lane & 16), 32);  // q broadcast
            se[u][r]  = qv - kk_t[r] + pe_t[r];
            vpe[t][r] = vv_t[r] + pe_t[r];
          }
        }
        v16h bs = packB_cd(se[0], se[1], hi);
        if (hh == 0) bs0 = bs; else bs1 = bs;
      }

      // h = relu(Wa1 @ s) : 16 x 16
      v8f h = wmma(ldfrag(frag, 44, lane), bs0, zero8());
      h     = wmma(ldfrag(frag, 45, lane), bs1, h);
#pragma unroll
      for (int r = 0; r < 8; ++r) h[r] = fmaxf(h[r], 0.f);

      // B fragment of h (K = 16 mid dims, padded to 32; hi lanes all zero)
      v8f xh = xchg16(h);
      v16h bh;
#pragma unroll
      for (int j = 0; j < 8; ++j) {
        bh[j]     = (_Float16)(hi ? 0.f : (float)h[j]);
        bh[j + 8] = (_Float16)(hi ? 0.f : (float)xh[j]);
      }

      // w = Wa2 @ h ; softmax over neighbors (lane butterfly) ; weighted sum
      v8f olv[4];
#pragma unroll
      for (int t = 0; t < 4; ++t) {
        v8f wl = wmma(ldfrag(frag, 46 + t, lane), bh, zero8());
#pragma unroll
        for (int r = 0; r < 8; ++r) {
          float mx  = rmax16(wl[r]);
          float e   = __expf(wl[r] - mx);
          float den = rsum16(e);
          float num = rsum16(e * vpe[t][r]);
          olv[t][r] = num / den;
        }
      }
      // stage this point's 64-vector (f16) for the batched projection
      if (ln < 8) {
        const int r = ln;
#pragma unroll
        for (int t = 0; t < 4; ++t)
          sto[i * OSTR + 16 * t + r + hi8] = (_Float16)pick(olv[t], r);
      }
    }

    // ---- Phase C: projection + residual + LayerNorm (lanes = point) -----
    {
      const _Float16* rp = sto + ln * OSTR;
      v16h b0 = bfrag_rowh(rp, hi, 0);
      v16h b1 = bfrag_rowh(rp, hi, 1);
      v8f o[4];
      float s1 = 0.f, s2 = 0.f;
#pragma unroll
      for (int t = 0; t < 4; ++t) {
        v8f a = wmma(ldfrag(frag, 32 + 2 * t, lane), b0, zero8());
        o[t]  = wmma(ldfrag(frag, 33 + 2 * t, lane), b1, a);
        v8f cf = ld8(cfeat + (size_t)(p0 + ln) * C + 16 * t + hi8);
#pragma unroll
        for (int r = 0; r < 8; ++r) {
          o[t][r] += cf[r];
          s1 += o[t][r];
          s2 += o[t][r] * o[t][r];
        }
      }
      s1 += __shfl_xor(s1, 16, 32);
      s2 += __shfl_xor(s2, 16, 32);
      const float mean = s1 * (1.f / 64.f);
      const float var  = s2 * (1.f / 64.f) - mean * mean;
      const float rstd = rsqrtf(var + EPS);
#pragma unroll
      for (int t = 0; t < 4; ++t) {
        v8f w8 = ld8(lnw + 16 * t + hi8);
        v8f b8 = ld8(lnb + 16 * t + hi8);
        v8f y;
#pragma unroll
        for (int r = 0; r < 8; ++r)
          y[r] = (o[t][r] - mean) * rstd * w8[r] + b8[r];
        st8(out + (size_t)(p0 + ln) * C + 16 * t + hi8, y);
      }
    }
  }
}

// ---------- host entry ----------------------------------------------------
extern "C" void kernel_launch(void* const* d_in, const int* in_sizes, int n_in,
                              void* d_out, int out_size, void* d_ws, size_t ws_size,
                              hipStream_t stream) {
  (void)in_sizes; (void)n_in; (void)out_size; (void)d_ws; (void)ws_size;
  const float* cxyz  = (const float*)d_in[0];
  const float* cfeat = (const float*)d_in[1];
  const float* nxyz  = (const float*)d_in[2];
  const float* nfeat = (const float*)d_in[3];
  const float* Wq    = (const float*)d_in[4];
  const float* Wk    = (const float*)d_in[5];
  const float* Wv    = (const float*)d_in[6];
  const float* Wpe1  = (const float*)d_in[7];
  const float* Wpe2  = (const float*)d_in[8];
  const float* Wa1   = (const float*)d_in[9];
  const float* Wa2   = (const float*)d_in[10];
  const float* Wproj = (const float*)d_in[11];
  const float* lnw   = (const float*)d_in[12];
  const float* lnb   = (const float*)d_in[13];
  float* out = (float*)d_out;

  const size_t smem = (size_t)FRAG_H * 2 + (size_t)NW * 16 * OSTR * 2;  // 60416 B
  dim3 grid(512), block(NW * 32);
  pt_attn_fused<<<grid, block, smem, stream>>>(cxyz, cfeat, nxyz, nfeat,
                                               Wq, Wk, Wv, Wpe1, Wpe2,
                                               Wa1, Wa2, Wproj, lnw, lnb, out);
}
